// DPLSTMModel_78005196030522
// MI455X (gfx1250) — matrix operand order repaired and checked by reading, not compiled, not run on hardware
//
#include <hip/hip_runtime.h>
#include <hip/hip_bf16.h>

// ---------------------------------------------------------------------------
// LSTM language model forward for MI455X (gfx1250).
// bf16 WMMA (v_wmma_f32_16x16x32_bf16) for all GEMMs, fp32 accumulation,
// TDM (tensor_load_to_lds) double-buffered B-tile staging in the GEMMs,
// persistent WMMA LSTM recurrence, fp32 log-softmax.
// Workspace requirement: ~162 MB.
// ---------------------------------------------------------------------------

typedef unsigned short u16;
typedef __attribute__((ext_vector_type(16))) __bf16 v16bf;
typedef __attribute__((ext_vector_type(8)))  __bf16 v8bf;
typedef __attribute__((ext_vector_type(8)))  float  v8f;
typedef __attribute__((ext_vector_type(4)))  unsigned v4u;
typedef __attribute__((ext_vector_type(8)))  int      v8i;
typedef __attribute__((ext_vector_type(4)))  int      v4i;

#define VOCAB 50257
#define HID   1024
#define BATCH 8
#define SEQT  256
#define MROWS (BATCH * SEQT)   // 2048
#define G4    (4 * HID)        // 4096

// float -> bf16, round-to-nearest-even
__device__ __forceinline__ u16 f32_to_bf16(float f) {
    unsigned u = __float_as_uint(f);
    unsigned r = 0x7FFFu + ((u >> 16) & 1u);
    u += r;
    return (u16)(u >> 16);
}

// LDS aperture keeps the LDS byte offset in addr[31:0] (ISA 10.2 aperture
// mapping) -> truncate the flat pointer.
__device__ __forceinline__ unsigned lds_byte_off(const void* p) {
    return (unsigned)(unsigned long long)p;
}

// ---------------------------------------------------------------------------
// TDM: async 2D tile load Global -> LDS (bf16 elements, data_size=2B).
// Descriptor per CDNA5 ISA 8.3/8.4:
//  group0: dw0 = count(1); dw1 = lds_addr; dw2/dw3 = global_addr[56:0], type=2
//  group1: data_size=1(2B); tensor_dim0/1 = remaining extents (OOB reads -> 0);
//          tile_dim0 x tile_dim1; tensor_dim0_stride (elements)
// Tile rows land contiguously in LDS: row n at lds_off + n*tile_d0*2 bytes.
// ---------------------------------------------------------------------------
__device__ __forceinline__ void tdm_load_2d_bf16(
    unsigned lds_off, const void* gptr,
    unsigned tile_d0, unsigned tile_d1,
    unsigned tensor_d0, unsigned tensor_d1, unsigned d0_stride) {
    unsigned long long ga = (unsigned long long)gptr;
    v4u g0;
    g0.x = 1u;                                        // count=1, user descriptor
    g0.y = lds_off;                                   // lds_addr (bytes)
    g0.z = (unsigned)(ga & 0xFFFFFFFFu);              // global_addr[31:0]
    g0.w = (unsigned)((ga >> 32) & 0x01FFFFFFu) | (2u << 30);  // [56:32], type=2
    v8i g1;
    g1[0] = (int)(1u << 16);                          // wg_mask=0, data_size=2B
    g1[1] = (int)((tensor_d0 & 0xFFFFu) << 16);       // tensor_dim0[15:0]
    g1[2] = (int)(((tensor_d0 >> 16) & 0xFFFFu) | ((tensor_d1 & 0xFFFFu) << 16));
    g1[3] = (int)(((tensor_d1 >> 16) & 0xFFFFu) | ((tile_d0 & 0xFFFFu) << 16));
    g1[4] = (int)(tile_d1 & 0xFFFFu);                 // tile_dim1, tile_dim2=0
    g1[5] = (int)d0_stride;                           // dim0 stride[31:0]
    g1[6] = 0;                                        // stride[47:32], dim1_stride
    g1[7] = 0;
    v4i z4 = {0, 0, 0, 0};
#if __clang_major__ >= 23
    v8i z8 = {0, 0, 0, 0, 0, 0, 0, 0};
    __builtin_amdgcn_tensor_load_to_lds(g0, g1, z4, z4, z8, 0);
#else
    __builtin_amdgcn_tensor_load_to_lds(g0, g1, z4, z4, 0);
#endif
}

// ---------------------------------------------------------------------------
// init: zero sync counter, h state (bf16), c state (f32)
// ---------------------------------------------------------------------------
__global__ void init_kernel(u16* __restrict__ h_bf, float* __restrict__ c_f32,
                            unsigned* __restrict__ cnt) {
    if (threadIdx.x == 0) *cnt = 0u;
    for (int i = threadIdx.x; i < BATCH * HID; i += 256) {
        h_bf[i] = 0;
        c_f32[i] = 0.0f;
    }
}

// ---------------------------------------------------------------------------
// generic fp32 -> bf16 conversion (grid-stride)
// ---------------------------------------------------------------------------
__global__ void cvt_kernel(const float* __restrict__ src, u16* __restrict__ dst,
                           size_t n) {
    size_t i = (size_t)blockIdx.x * 256 + threadIdx.x;
    size_t stride = (size_t)gridDim.x * 256;
    for (; i < n; i += stride) dst[i] = f32_to_bf16(src[i]);
}

// ---------------------------------------------------------------------------
// embedding gather + bf16 convert: row r = b*T + t
// ---------------------------------------------------------------------------
__global__ void gather_kernel(const int* __restrict__ x,
                              const float* __restrict__ embW,
                              u16* __restrict__ emb_bf) {
    int row = blockIdx.x;
    int tok = x[row];
    const float* src = embW + (size_t)tok * HID;
    u16* dst = emb_bf + (size_t)row * HID;
    for (int i = threadIdx.x; i < HID; i += 256) dst[i] = f32_to_bf16(src[i]);
}

// ---------------------------------------------------------------------------
// Tiled bf16 WMMA GEMM:  C[M,N] = A[M,K] * B[N,K]^T + bias0[n] + bias1[n]
// A,B row-major bf16 (K contiguous). Block tile 128x128, 8 waves,
// wave v owns rows [16v,16v+16) x all 8 N-subtiles.
// B tile (128 rows x 32 k, 8 KB) staged by the TDM, double-buffered:
// wave 0 issues tensor_load_to_lds for tile k+1, then s_wait_tensorcnt for
// tile k, so DMA overlaps the 8 WMMAs of the current tile.
//
// Fragment layouts used here (A/B in contiguous k order, not ISA table order):
//   A 16x32 bf16: lane l -> m = l&15, khalf g = l>>4;
//                 elems e -> k = kb + 16g + e
//   B 32x16 bf16: lane l -> n = l&15, kh = l>>4; elems e -> k = kb + 16*kh + e
//   C 16x16 f32:  lane l -> n = l&15; elem j -> m = j + 8*(l>>4)
// ---------------------------------------------------------------------------
__global__ __launch_bounds__(256) void gemm_bf16_kernel(
    const u16* __restrict__ A, const u16* __restrict__ B,
    float* __restrict__ C,
    const float* __restrict__ bias0, const float* __restrict__ bias1,
    int M, int N, int K, int ldc) {
    __shared__ __align__(32) u16 sB[2][128 * 32];   // double-buffered, 16 KB

    const int mb = blockIdx.x * 128;
    const int nb = blockIdx.y * 128;
    const int lane = threadIdx.x & 31;
    const int wave = threadIdx.x >> 5;
    const int n16 = lane & 15;
    const int g   = lane >> 4;

    v8f acc[8] = {};  // 8 N-subtiles

    const u16* arow = A + (size_t)(mb + wave * 16 + n16) * K;  // this lane's A row
    const u16* btile = B + (size_t)nb * K;                     // B tile row start
    const unsigned nrem = (unsigned)(N - nb);                  // rows remaining
    const bool issuer = (wave == 0);

    if (issuer) {
        tdm_load_2d_bf16(lds_byte_off(&sB[0][0]), btile,
                         32u, 128u, (unsigned)K, nrem, (unsigned)K);
    }

    int cur = 0;
    for (int kb = 0; kb < K; kb += 32, cur ^= 1) {
        if (issuer) {
            if (kb + 32 < K) {
                tdm_load_2d_bf16(lds_byte_off(&sB[cur ^ 1][0]), btile + kb + 32,
                                 32u, 128u, (unsigned)(K - kb - 32), nrem,
                                 (unsigned)K);
                __builtin_amdgcn_s_wait_tensorcnt(1);   // current tile done
            } else {
                __builtin_amdgcn_s_wait_tensorcnt(0);
            }
        }
        __syncthreads();                                 // tile visible to WG

        // ---- A fragment (global) ----
        union { v16bf v; v8bf h8[2]; } af;
        af.h8[0] = *reinterpret_cast<const v8bf*>(arow + kb + 16 * g);
        af.h8[1] = *reinterpret_cast<const v8bf*>(arow + kb + 16 * g + 8);

        // ---- 8 WMMAs over the N-subtiles (B from LDS) ----
#pragma unroll
        for (int sub = 0; sub < 8; ++sub) {
            const v16bf bf = *reinterpret_cast<const v16bf*>(
                &sB[cur][(sub * 16 + n16) * 32 + 16 * g]);
            acc[sub] = __builtin_amdgcn_wmma_f32_16x16x32_bf16(
                false, af.v, false, bf, (short)0, acc[sub], false, false);
        }
        __syncthreads();                                 // reads done before reuse
    }

    // ---- store C (+bias), guard N ----
#pragma unroll
    for (int sub = 0; sub < 8; ++sub) {
        int col = nb + sub * 16 + n16;
        if (col < N) {
            float bv = (bias0 ? bias0[col] : 0.0f) + (bias1 ? bias1[col] : 0.0f);
#pragma unroll
            for (int e = 0; e < 8; ++e) {
                int row = mb + wave * 16 + e + 8 * g;
                C[(size_t)row * ldc + col] = acc[sub][e] + bv;
            }
        }
    }
}

// ---------------------------------------------------------------------------
// Persistent LSTM recurrence. 32 WGs x 256 threads (8 waves).
// WG w owns hidden units u in [32w, 32w+32); wave v -> gate quadrant q = v>>1,
// sub-tile s = v&1, columns cb = q*1024 + 32w + 16s .. +15.
// Per step: gates = x_proj[t] + h @ w_hh^T (WMMA, batch 8 padded to M=16),
// gate math in fp32, h written as bf16 (next step A + decoder input),
// then device-wide barrier via atomic counter.
// ---------------------------------------------------------------------------
#define LSTM_WGS 32

__global__ __launch_bounds__(256) void lstm_kernel(
    const float* __restrict__ x_proj,    // [(b*T+t), 4096]
    const u16*   __restrict__ whh_bf,    // [4096, 1024]
    u16*         __restrict__ h_bf,      // [8, 1024]
    u16*         __restrict__ out_bf,    // [(b*T+t), 1024]
    float*       __restrict__ c_f32,     // [8, 1024]
    float*       __restrict__ h_last,    // [8, 1024]
    unsigned*    __restrict__ sync_cnt) {
    __shared__ float sg[4 * 32 * 8];     // [quad][unit][b]

    const int wg   = blockIdx.x;
    const int lane = threadIdx.x & 31;
    const int wave = threadIdx.x >> 5;
    const int quad = wave >> 1;
    const int sub  = wave & 1;
    const int cb   = quad * HID + wg * 32 + sub * 16;
    const int n16  = lane & 15;
    const int g    = lane >> 4;

    const u16* brow = whh_bf + (size_t)(cb + n16) * HID + 16 * g;
    const u16* arow = h_bf + (size_t)n16 * HID;   // valid only for n16 < 8
    const bool arow_ok = (n16 < 8);

    const int pb = threadIdx.x >> 5;     // phase-2 batch index 0..7
    const int ul = threadIdx.x & 31;     // phase-2 local unit 0..31
    const int ug = wg * 32 + ul;         // global hidden unit

    for (int t = 0; t < SEQT; ++t) {
        // ---- phase 1: gate pre-activations via WMMA ----
        v8f acc = {};
#pragma unroll 4
        for (int kb = 0; kb < HID; kb += 32) {
            union { v16bf v; v8bf h8[2]; unsigned u[8]; } af;
            if (arow_ok) {
                af.h8[0] = *reinterpret_cast<const v8bf*>(arow + kb + 16 * g);
                af.h8[1] = *reinterpret_cast<const v8bf*>(arow + kb + 16 * g + 8);
            } else {
#pragma unroll
                for (int q = 0; q < 8; ++q) af.u[q] = 0u;   // pad rows 8..15
            }
            const v16bf bf = *reinterpret_cast<const v16bf*>(brow + kb);
            acc = __builtin_amdgcn_wmma_f32_16x16x32_bf16(
                false, af.v, false, bf, (short)0, acc, false, false);
        }
        if (g == 0) {   // lanes 0..15 hold batch rows 0..7 in elems 0..7
#pragma unroll
            for (int b = 0; b < BATCH; ++b) {
                float v = acc[b] +
                    x_proj[(size_t)(b * SEQT + t) * G4 + cb + n16];
                sg[(quad * 32 + sub * 16 + n16) * 8 + b] = v;
            }
        }
        __syncthreads();

        // ---- phase 2: activations + state update (one (b,u) per thread) ----
        {
            float gi = sg[(0 * 32 + ul) * 8 + pb];
            float gf = sg[(1 * 32 + ul) * 8 + pb];
            float gg = sg[(2 * 32 + ul) * 8 + pb];
            float go = sg[(3 * 32 + ul) * 8 + pb];
            gi = 1.0f / (1.0f + __expf(-gi));
            gf = 1.0f / (1.0f + __expf(-gf));
            go = 1.0f / (1.0f + __expf(-go));
            gg = tanhf(gg);
            float c = gf * c_f32[pb * HID + ug] + gi * gg;
            c_f32[pb * HID + ug] = c;
            float h = go * tanhf(c);
            u16 hb = f32_to_bf16(h);
            h_bf[pb * HID + ug] = hb;
            out_bf[(size_t)(pb * SEQT + t) * HID + ug] = hb;
            if (t == SEQT - 1) h_last[pb * HID + ug] = h;
        }
        __syncthreads();

        // ---- device-wide barrier ----
        if (threadIdx.x == 0) {
            __threadfence();
            __hip_atomic_fetch_add(sync_cnt, 1u, __ATOMIC_ACQ_REL,
                                   __HIP_MEMORY_SCOPE_AGENT);
            unsigned target = (unsigned)LSTM_WGS * (unsigned)(t + 1);
            while (__hip_atomic_load(sync_cnt, __ATOMIC_ACQUIRE,
                                     __HIP_MEMORY_SCOPE_AGENT) < target) {
                __builtin_amdgcn_s_sleep(2);
            }
            __threadfence();
        }
        __syncthreads();
    }
}

// ---------------------------------------------------------------------------
// row-wise log-softmax, in place. One block per row, 256 threads.
// ---------------------------------------------------------------------------
__global__ __launch_bounds__(256) void logsoftmax_kernel(float* __restrict__ p0,
                                                         int V) {
    float* p = p0 + (size_t)blockIdx.x * V;
    __shared__ float red[8];

    float m = -3.4e38f;
    for (int i = threadIdx.x; i < V; i += 256) m = fmaxf(m, p[i]);
#pragma unroll
    for (int off = 16; off; off >>= 1) m = fmaxf(m, __shfl_xor(m, off, 32));
    if ((threadIdx.x & 31) == 0) red[threadIdx.x >> 5] = m;
    __syncthreads();
    if (threadIdx.x == 0) {
        float mm = red[0];
        for (int i = 1; i < 8; ++i) mm = fmaxf(mm, red[i]);
        red[0] = mm;
    }
    __syncthreads();
    const float rowmax = red[0];
    __syncthreads();

    float s = 0.0f;
    for (int i = threadIdx.x; i < V; i += 256) s += __expf(p[i] - rowmax);
#pragma unroll
    for (int off = 16; off; off >>= 1) s += __shfl_xor(s, off, 32);
    if ((threadIdx.x & 31) == 0) red[threadIdx.x >> 5] = s;
    __syncthreads();
    if (threadIdx.x == 0) {
        float ss = 0.0f;
        for (int i = 0; i < 8; ++i) ss += red[i];
        red[0] = logf(ss);
    }
    __syncthreads();
    const float lse = red[0];

    for (int i = threadIdx.x; i < V; i += 256) p[i] = p[i] - rowmax - lse;
}

// ---------------------------------------------------------------------------
// copy final (h, c) into tail of d_out
// ---------------------------------------------------------------------------
__global__ void copy_hidden_kernel(const float* __restrict__ h,
                                   const float* __restrict__ c,
                                   float* __restrict__ dst) {
    int i = blockIdx.x * 256 + threadIdx.x;
    if (i < BATCH * HID) dst[i] = h[i];
    else                 dst[i] = c[i - BATCH * HID];
}

// ---------------------------------------------------------------------------
// host side
// ---------------------------------------------------------------------------
extern "C" void kernel_launch(void* const* d_in, const int* in_sizes, int n_in,
                              void* d_out, int out_size, void* d_ws, size_t ws_size,
                              hipStream_t stream) {
    const int*   x     = (const int*)  d_in[0];   // [8,256]
    const float* emb_W = (const float*)d_in[1];   // [50257,1024]
    const float* w_ih  = (const float*)d_in[2];   // [4096,1024]
    const float* w_hh  = (const float*)d_in[3];   // [4096,1024]
    const float* b_ih  = (const float*)d_in[4];   // [4096]
    const float* b_hh  = (const float*)d_in[5];   // [4096]
    const float* dec_b = (const float*)d_in[6];   // [50257]
    (void)in_sizes; (void)n_in; (void)out_size; (void)ws_size;

    // workspace layout (256-byte aligned slices)
    char* ws = (char*)d_ws;
    size_t off = 0;
    auto take = [&](size_t bytes) {
        char* p = ws + off;
        off = (off + bytes + 255) & ~(size_t)255;
        return p;
    };
    u16*      embW_bf = (u16*)     take((size_t)VOCAB * HID * 2);   // 103 MB
    u16*      wih_bf  = (u16*)     take((size_t)G4 * HID * 2);      // 8 MB
    u16*      whh_bf  = (u16*)     take((size_t)G4 * HID * 2);      // 8 MB
    u16*      emb_bf  = (u16*)     take((size_t)MROWS * HID * 2);   // 4 MB
    float*    x_proj  = (float*)   take((size_t)MROWS * G4 * 4);    // 33.5 MB
    u16*      out_bf  = (u16*)     take((size_t)MROWS * HID * 2);   // 4 MB
    u16*      h_bf    = (u16*)     take((size_t)BATCH * HID * 2);
    float*    c_f32   = (float*)   take((size_t)BATCH * HID * 4);
    float*    h_last  = (float*)   take((size_t)BATCH * HID * 4);
    unsigned* cnt     = (unsigned*)take(256);

    float* logits = (float*)d_out;                               // [2048,50257]
    float* hidden = logits + (size_t)MROWS * VOCAB;              // h then c

    // 1. init state + sync counter (must reset every launch: deterministic)
    init_kernel<<<1, 256, 0, stream>>>(h_bf, c_f32, cnt);

    // 2. fp32 -> bf16 conversions
    cvt_kernel<<<8192, 256, 0, stream>>>(emb_W, embW_bf, (size_t)VOCAB * HID);
    cvt_kernel<<<4096, 256, 0, stream>>>(w_ih, wih_bf, (size_t)G4 * HID);
    cvt_kernel<<<4096, 256, 0, stream>>>(w_hh, whh_bf, (size_t)G4 * HID);

    // 3. embedding gather
    gather_kernel<<<MROWS, 256, 0, stream>>>(x, emb_W, emb_bf);

    // 4. x_proj = emb @ w_ih^T + (b_ih + b_hh)   [2048 x 4096]
    gemm_bf16_kernel<<<dim3(MROWS / 128, G4 / 128), 256, 0, stream>>>(
        emb_bf, wih_bf, x_proj, b_ih, b_hh, MROWS, G4, HID, G4);

    // 5. LSTM recurrence (persistent, device-wide sync per step)
    lstm_kernel<<<LSTM_WGS, 256, 0, stream>>>(x_proj, whh_bf, h_bf, out_bf,
                                              c_f32, h_last, cnt);

    // 6. decoder: logits = out @ emb_W^T + dec_b   [2048 x 50257]
    gemm_bf16_kernel<<<dim3(MROWS / 128, (VOCAB + 127) / 128), 256, 0, stream>>>(
        out_bf, embW_bf, logits, dec_b, nullptr, MROWS, VOCAB, HID, VOCAB);

    // 7. log-softmax in place
    logsoftmax_kernel<<<MROWS, 256, 0, stream>>>(logits, VOCAB);

    // 8. final hidden states
    copy_hidden_kernel<<<(2 * BATCH * HID) / 256, 256, 0, stream>>>(
        h_last, c_f32, hidden);
}
